// BasicBlock_90589450207910
// MI455X (gfx1250) — 8-bit matrix operand order repaired and checked by reading, compiled, not run on hardware
//
#include <hip/hip_runtime.h>
#include <stdint.h>

typedef int v8i __attribute__((ext_vector_type(8)));

static constexpr int kB  = 64;
static constexpr int kC  = 256;
static constexpr int kH  = 28;
static constexpr int kW  = 28;
static constexpr int kHW = kH * kW;
static constexpr int kTotal = kB * kC * kHW;
static constexpr float kScale = 0.1f;
static constexpr float kBnEps = 1e-5f;

__device__ __forceinline__ int8_t fsign8(float v) {
    return (int8_t)((v > 0.f) - (v < 0.f));
}

__global__ void pack_weights(const float* __restrict__ w, int8_t* __restrict__ wf) {
    int gid = blockIdx.x * blockDim.x + threadIdx.x;
    if (gid >= 9 * 4 * 16 * 32) return;
    int lane = gid & 31;
    int frag = gid >> 5;
    int nt   = frag & 15;
    int kcf  = frag >> 4;
    int kc   = kcf & 3;
    int tap  = kcf >> 2;
    int kh = tap / 3, kw = tap % 3;
    int cout = nt * 16 + (lane & 15);
    int kbase = (lane >> 4) * 16;

    int words[8];
#pragma unroll
    for (int v = 0; v < 8; ++v) {
        int wrd = 0;
#pragma unroll
        for (int bb = 0; bb < 4; ++bb) {
            int k = (v < 4 ? v * 4 + bb : 32 + (v - 4) * 4 + bb) + kbase;
            int cin = kc * 64 + k;
            float wv = w[((size_t)(cout * kC + cin) * 3 + kh) * 3 + kw];
            wrd |= ((int)(uint8_t)fsign8(wv)) << (8 * bb);
        }
        words[v] = wrd;
    }
    v8i out = { words[0], words[1], words[2], words[3],
                words[4], words[5], words[6], words[7] };
    *(v8i*)(wf + (size_t)gid * 32) = out;
}

__global__ void calc_alpha(const float* __restrict__ w, float* __restrict__ sa) {
    int c = blockIdx.x;
    int t = threadIdx.x;
    __shared__ float red[256];
    float s = 0.f;
    for (int i = t; i < kC * 9; i += 256) s += fabsf(w[(size_t)c * kC * 9 + i]);
    red[t] = s;
    __syncthreads();
    for (int o = 128; o > 0; o >>= 1) {
        if (t < o) red[t] += red[t + o];
        __syncthreads();
    }
    if (t == 0) sa[c] = kScale * red[0] / (float)(kC * 9);
}

__global__ void binarize_in(const float* __restrict__ x, int8_t* __restrict__ abin) {
    for (int i = blockIdx.x * blockDim.x + threadIdx.x; i < kTotal;
         i += gridDim.x * blockDim.x) {
        int hw = i % kHW;
        int c  = (i / kHW) % kC;
        int nb = i / (kC * kHW);
        abin[((size_t)nb * kHW + hw) * kC + c] = fsign8(x[i]);
    }
}

__global__ void __launch_bounds__(256)
conv_wmma(const int8_t* __restrict__ abin, const int8_t* __restrict__ wf,
          const float* __restrict__ sa, float* __restrict__ y) {
    const int tid   = threadIdx.x;
    const int lane  = tid & 31;
    const int wave  = tid >> 5;
    const int mg    = wave >> 2;
    const int cg    = wave & 3;
    const int lhalf = lane >> 4;
    const int l15   = lane & 15;
    const int mbase = blockIdx.x * 8 + mg * 4;

    int nbm[4], ohm[4], owm[4];
#pragma unroll
    for (int m = 0; m < 4; ++m) {
        int Mi = (mbase + m) * 16 + l15;
        nbm[m] = Mi / kHW;
        int p  = Mi % kHW;
        ohm[m] = p / kW;
        owm[m] = p % kW;
    }

    v8i acc[4][4];
#pragma unroll
    for (int m = 0; m < 4; ++m)
#pragma unroll
        for (int j = 0; j < 4; ++j)
            acc[m][j] = (v8i){0, 0, 0, 0, 0, 0, 0, 0};

#pragma unroll 1
    for (int tap = 0; tap < 9; ++tap) {
        const int dh = tap / 3 - 1;
        const int dw = tap % 3 - 1;
        size_t abase[4];
        bool   inb[4];
#pragma unroll
        for (int m = 0; m < 4; ++m) {
            int ih = ohm[m] + dh;
            int iw = owm[m] + dw;
            inb[m]   = ((unsigned)ih < (unsigned)kH) && ((unsigned)iw < (unsigned)kW);
            abase[m] = inb[m] ? ((size_t)((nbm[m] * kH + ih) * kW + iw)) * kC : 0;
        }
#pragma unroll
        for (int kc = 0; kc < 4; ++kc) {
            const int8_t* wbase =
                wf + ((size_t)(((tap * 4 + kc) * 16 + cg * 4) * 32 + lane)) * 32;
            __builtin_prefetch(wbase + 16384, 0, 0);
            v8i b0 = *(const v8i*)(wbase);
            v8i b1 = *(const v8i*)(wbase + 1024);
            v8i b2 = *(const v8i*)(wbase + 2048);
            v8i b3 = *(const v8i*)(wbase + 3072);
#pragma unroll
            for (int m = 0; m < 4; ++m) {
                v8i a;
#pragma unroll
                for (int i2 = 0; i2 < 4; ++i2) {
                    uint64_t u = 0;
                    if (inb[m])
                        u = *(const uint64_t*)(abin + abase[m] + kc * 64 + (i2 >> 1) * 32 +
                                               lhalf * 16 + (i2 & 1) * 8);
                    a[2 * i2]     = (int)(uint32_t)u;
                    a[2 * i2 + 1] = (int)(uint32_t)(u >> 32);
                }
                acc[m][0] = __builtin_amdgcn_wmma_i32_16x16x64_iu8(true, a, true, b0,
                                                                   acc[m][0], false, false);
                acc[m][1] = __builtin_amdgcn_wmma_i32_16x16x64_iu8(true, a, true, b1,
                                                                   acc[m][1], false, false);
                acc[m][2] = __builtin_amdgcn_wmma_i32_16x16x64_iu8(true, a, true, b2,
                                                                   acc[m][2], false, false);
                acc[m][3] = __builtin_amdgcn_wmma_i32_16x16x64_iu8(true, a, true, b3,
                                                                   acc[m][3], false, false);
            }
        }
    }

    const int row0 = lhalf * 8;
#pragma unroll
    for (int m = 0; m < 4; ++m) {
        union { v8i v; int s[8]; } ua[4];
        ua[0].v = acc[m][0];
        ua[1].v = acc[m][1];
        ua[2].v = acc[m][2];
        ua[3].v = acc[m][3];
        const int Mo_base = (mbase + m) * 16 + row0;
#pragma unroll
        for (int j = 0; j < 4; ++j) {
            const int cout = cg * 64 + j * 16 + l15;
            const float s = sa[cout];
#pragma unroll
            for (int r = 0; r < 8; ++r) {
                const int Mo  = Mo_base + r;
                const int nb2 = Mo / kHW;
                const int p2  = Mo % kHW;
                y[((size_t)(nb2 * kC + cout)) * kHW + p2] = (float)ua[j].s[r] * s;
            }
        }
    }
}

__global__ void chan_stats(const float* __restrict__ y, float* __restrict__ mu,
                           float* __restrict__ var) {
    int c = blockIdx.x;
    int t = threadIdx.x;
    __shared__ float rs[256];
    __shared__ float rq[256];
    float s = 0.f, q = 0.f;
    for (int nb = 0; nb < kB; ++nb) {
        const float* row = y + ((size_t)(nb * kC + c)) * kHW;
        for (int i = t; i < kHW; i += 256) {
            float v = row[i];
            s += v;
            q += v * v;
        }
    }
    rs[t] = s; rq[t] = q;
    __syncthreads();
    for (int o = 128; o > 0; o >>= 1) {
        if (t < o) { rs[t] += rs[t + o]; rq[t] += rq[t + o]; }
        __syncthreads();
    }
    if (t == 0) {
        float inv = 1.f / (float)(kB * kHW);
        float m = rs[0] * inv;
        mu[c]  = m;
        var[c] = rq[0] * inv - m * m;
    }
}

__global__ void fuse_bn_res_prelu(const float* __restrict__ y,
                                  const float* __restrict__ res,
                                  const float* __restrict__ mu,
                                  const float* __restrict__ var,
                                  const float* __restrict__ gamma,
                                  const float* __restrict__ beta,
                                  const float* __restrict__ pa,
                                  float* __restrict__ outf,
                                  int8_t* __restrict__ abin) {
    const float a = pa[0];
    for (int i = blockIdx.x * blockDim.x + threadIdx.x; i < kTotal;
         i += gridDim.x * blockDim.x) {
        int hw = i % kHW;
        int c  = (i / kHW) % kC;
        int nb = i / (kC * kHW);
        float v = (y[i] - mu[c]) * rsqrtf(var[c] + kBnEps) * gamma[c] + beta[c] + res[i];
        v = (v >= 0.f) ? v : a * v;
        outf[i] = v;
        if (abin)
            abin[((size_t)nb * kHW + hw) * kC + c] = fsign8(v);
    }
}

extern "C" void kernel_launch(void* const* d_in, const int* in_sizes, int n_in,
                              void* d_out, int out_size, void* d_ws, size_t ws_size,
                              hipStream_t stream) {
    (void)in_sizes; (void)n_in; (void)out_size; (void)ws_size;
    const float* x   = (const float*)d_in[0];
    const float* w1  = (const float*)d_in[1];
    const float* w2  = (const float*)d_in[2];
    const float* g1  = (const float*)d_in[3];
    const float* be1 = (const float*)d_in[4];
    const float* g2  = (const float*)d_in[5];
    const float* be2 = (const float*)d_in[6];
    const float* pa1 = (const float*)d_in[7];
    const float* pa2 = (const float*)d_in[8];
    float* out = (float*)d_out;

    uint8_t* ws = (uint8_t*)d_ws;
    size_t off = 0;
    int8_t* abin = (int8_t*)(ws + off); off += (size_t)kTotal;
    float*  y    = (float*)(ws + off);  off += (size_t)kTotal * 4;
    int8_t* wf1  = (int8_t*)(ws + off); off += 9 * 4 * 16 * 32 * 32;
    int8_t* wf2  = (int8_t*)(ws + off); off += 9 * 4 * 16 * 32 * 32;
    float*  sa1  = (float*)(ws + off);  off += 1024;
    float*  sa2  = (float*)(ws + off);  off += 1024;
    float*  mu   = (float*)(ws + off);  off += 1024;
    float*  var  = (float*)(ws + off);  off += 1024;

    pack_weights<<<72, 256, 0, stream>>>(w1, wf1);
    pack_weights<<<72, 256, 0, stream>>>(w2, wf2);
    calc_alpha<<<256, 256, 0, stream>>>(w1, sa1);
    calc_alpha<<<256, 256, 0, stream>>>(w2, sa2);

    binarize_in<<<4096, 256, 0, stream>>>(x, abin);
    conv_wmma<<<392, 256, 0, stream>>>(abin, wf1, sa1, y);
    chan_stats<<<256, 256, 0, stream>>>(y, mu, var);
    fuse_bn_res_prelu<<<4096, 256, 0, stream>>>(y, x, mu, var, g1, be1, pa1, out, abin);

    conv_wmma<<<392, 256, 0, stream>>>(abin, wf2, sa2, y);
    chan_stats<<<256, 256, 0, stream>>>(y, mu, var);
    fuse_bn_res_prelu<<<4096, 256, 0, stream>>>(y, out, mu, var, g2, be2, pa2, out, nullptr);
}
